// RNNTagger_15564961480882
// MI455X (gfx1250) — matrix operand order repaired and checked by reading, not compiled, not run on hardware
//
#include <hip/hip_runtime.h>
#include <hip/hip_bf16.h>
#include <math.h>

typedef __attribute__((ext_vector_type(16))) __bf16 v16bf;
typedef __attribute__((ext_vector_type(8)))  __bf16 bf16x8;
typedef __attribute__((ext_vector_type(8)))  float  v8f;

#define BB   64
#define TT   512
#define NTOK (BB*TT)     // 32768
#define EE   256
#define HH   600
#define GDIM 2400        // 4*H, unit-major permuted gate columns

// ---------------------------------------------------------------------------
// Staging fetch: each thread owns one 16B A-chunk and two 16B B-chunks of the
// 64x32 (A) / 128x32 (B) K-tile. All guards fold away unless this is the
// K-tail tile (K%32 != 0) or an EDGE block (partial N).
// ---------------------------------------------------------------------------
template<int K, bool EDGE>
__device__ __forceinline__ void stage_fetch(
    const __bf16* __restrict__ A, const __bf16* __restrict__ Bw,
    int kb, int Ncols, int mblk, int nblk,
    int ar, int ach, int br0, int br1, int bch,
    bf16x8& va, bf16x8& vb0, bf16x8& vb1)
{
  constexpr int KT_FULL = K / 32;
  constexpr int KTAIL   = K % 32;          // multiple of 8 for all our shapes
  const int k0 = kb * 32;
  const bool tail = KTAIL && (kb == KT_FULL);
  bf16x8 z = {};
  va = z; vb0 = z; vb1 = z;
  const bool ok_a = !tail || (ach + 8 <= KTAIL);
  const bool ok_b = !tail || (bch + 8 <= KTAIL);
  if (ok_a)
    va = *(const bf16x8*)(A + (size_t)(mblk + ar) * K + k0 + ach);
  if (ok_b && (!EDGE || nblk + br0 < Ncols))
    vb0 = *(const bf16x8*)(Bw + (size_t)(nblk + br0) * K + k0 + bch);
  if (ok_b && (!EDGE || nblk + br1 < Ncols))
    vb1 = *(const bf16x8*)(Bw + (size_t)(nblk + br1) * K + k0 + bch);
}

// 4 WMMAs per wave on one staged K-tile: wave owns M-tile mt, N-half nh.
__device__ __forceinline__ void wmma_tiles(
    const __bf16 (*__restrict__ As)[32], const __bf16 (*__restrict__ Bs)[32],
    int lane, int mt, int nh, v8f acc[4])
{
  const int ksel = lane >> 4;
  // A fragment: lane L (m=L&15): K = {ksel*8..+7} U {ksel*8+16..+23}
  const __bf16* arow = &As[mt*16 + (lane & 15)][ksel * 8];
  bf16x8 a0 = *(const bf16x8*)(arow);
  bf16x8 a1 = *(const bf16x8*)(arow + 16);
  v16bf af = __builtin_shufflevector(a0, a1,
      0,1,2,3,4,5,6,7,8,9,10,11,12,13,14,15);
#pragma unroll
  for (int nt = 0; nt < 4; ++nt) {
    // B fragment: lane L: column n=L&15, K = {ksel*8..+7} U {ksel*8+16..+23}
    const __bf16* brow = &Bs[nh*64 + nt*16 + (lane & 15)][ksel * 8];
    bf16x8 b0 = *(const bf16x8*)(brow);
    bf16x8 b1 = *(const bf16x8*)(brow + 16);
    v16bf bf = __builtin_shufflevector(b0, b1,
        0,1,2,3,4,5,6,7,8,9,10,11,12,13,14,15);
    acc[nt] = __builtin_amdgcn_wmma_f32_16x16x32_bf16(
        false, af, false, bf, (short)0, acc[nt], false, false);
  }
}

// ---------------------------------------------------------------------------
// bf16 GEMM: C[n,j] = sum_k A[n,k]*Bw[j,k] + bias[j]
// Double-buffered LDS pipeline: one barrier per K-tile; next tile is fetched
// into registers while WMMAs consume the current LDS tile.
// ---------------------------------------------------------------------------
template<int K, bool EDGE>
__global__ __launch_bounds__(256) void gemm_bf16_t(
    const __bf16* __restrict__ A, const __bf16* __restrict__ Bw,
    const float* __restrict__ bias, float* __restrict__ C,
    int Ncols, int nblkBase)
{
  __shared__ __bf16 As[2][64][32];
  __shared__ __bf16 Bs[2][128][32];
  __shared__ float  sg[64][128];
  const int tid = threadIdx.x, lane = tid & 31, wave = tid >> 5;
  const int mt = wave & 3, nh = wave >> 2;
  const int mblk = blockIdx.x * 64;
  const int nblk = nblkBase + blockIdx.y * 128;
  const int ar = tid >> 2, ach = (tid & 3) * 8;
  const int br0 = tid >> 2, br1 = br0 + 64, bch = (tid & 3) * 8;

  constexpr int NT = K / 32 + ((K % 32) ? 1 : 0);
  v8f acc[4] = {};
  bf16x8 ra, rb0, rb1;
  stage_fetch<K, EDGE>(A, Bw, 0, Ncols, mblk, nblk, ar, ach, br0, br1, bch, ra, rb0, rb1);
  *(bf16x8*)(&As[0][ar][ach])  = ra;
  *(bf16x8*)(&Bs[0][br0][bch]) = rb0;
  *(bf16x8*)(&Bs[0][br1][bch]) = rb1;
  __syncthreads();
  for (int kb = 0; kb < NT; ++kb) {
    const int cur = kb & 1;
    if (kb + 1 < NT)
      stage_fetch<K, EDGE>(A, Bw, kb+1, Ncols, mblk, nblk, ar, ach, br0, br1, bch, ra, rb0, rb1);
    wmma_tiles(As[cur], Bs[cur], lane, mt, nh, acc);
    if (kb + 1 < NT) {
      *(bf16x8*)(&As[cur^1][ar][ach])  = ra;
      *(bf16x8*)(&Bs[cur^1][br0][bch]) = rb0;
      *(bf16x8*)(&Bs[cur^1][br1][bch]) = rb1;
    }
    __syncthreads();
  }
  // epilogue through LDS -> coalesced float4 stores
  // C layout: lane L: col = L&15, rows = vgpr r + 8*(L>>4)
#pragma unroll
  for (int nt = 0; nt < 4; ++nt)
#pragma unroll
    for (int r = 0; r < 8; ++r)
      sg[mt*16 + ((lane >> 4) << 3) + r][nh*64 + nt*16 + (lane & 15)] = acc[nt][r];
  __syncthreads();
  for (int i = tid; i < 64*32; i += 256) {
    int r = i >> 5, cch = (i & 31) * 4;
    int col = nblk + cch;
    if (!EDGE || col < Ncols) {
      float4 v = *(const float4*)&sg[r][cch];
      if (bias) {
        float4 bv = *(const float4*)&bias[col];
        v.x += bv.x; v.y += bv.y; v.z += bv.z; v.w += bv.w;
      }
      *(float4*)&C[(size_t)(mblk + r) * Ncols + col] = v;
    }
  }
}

// ---------------------------------------------------------------------------
// One LSTM timestep, both directions fused (blockIdx.y = dir).
// h_prev @ Whh^T via WMMA (K=600 fixed, same double-buffered pipeline), plus
// precomputed G = x@Wih^T + b (gate-permuted unit-major), LSTM cell epilogue.
// ---------------------------------------------------------------------------
template<bool EDGE>
__global__ __launch_bounds__(256) void lstm_step_t(
    const __bf16* __restrict__ hinF, const __bf16* __restrict__ hinB,
    __bf16* __restrict__ hnextF, __bf16* __restrict__ hnextB,
    const __bf16* __restrict__ WhhF, const __bf16* __restrict__ WhhB,
    const float* __restrict__ GF, const float* __restrict__ GB,
    float* __restrict__ cF, float* __restrict__ cB,
    float* __restrict__ hlayer, int t, int nblkBase)
{
  __shared__ __bf16 As[2][64][32];
  __shared__ __bf16 Bs[2][128][32];
  __shared__ float  sg[64][128];

  const int dir = blockIdx.y;
  const __bf16* hin   = dir ? hinB : hinF;
  __bf16*       hnext = dir ? hnextB : hnextF;
  const __bf16* Whh   = dir ? WhhB : WhhF;
  const float*  G     = dir ? GB : GF;
  float*        cst   = dir ? cB : cF;
  const int time = dir ? (TT - 1 - t) : t;
  const int nblk = nblkBase + blockIdx.x * 128;

  const int tid = threadIdx.x, lane = tid & 31, wave = tid >> 5;
  const int mt = wave & 3, nh = wave >> 2;
  const int ar = tid >> 2, ach = (tid & 3) * 8;
  const int br0 = tid >> 2, br1 = br0 + 64, bch = (tid & 3) * 8;

  constexpr int K  = HH;                         // 600
  constexpr int NT = K / 32 + ((K % 32) ? 1 : 0); // 19
  v8f acc[4] = {};
  bf16x8 ra, rb0, rb1;
  stage_fetch<K, EDGE>(hin, Whh, 0, GDIM, 0, nblk, ar, ach, br0, br1, bch, ra, rb0, rb1);
  *(bf16x8*)(&As[0][ar][ach])  = ra;
  *(bf16x8*)(&Bs[0][br0][bch]) = rb0;
  *(bf16x8*)(&Bs[0][br1][bch]) = rb1;
  __syncthreads();
  for (int kb = 0; kb < NT; ++kb) {
    const int cur = kb & 1;
    if (kb + 1 < NT)
      stage_fetch<K, EDGE>(hin, Whh, kb+1, GDIM, 0, nblk, ar, ach, br0, br1, bch, ra, rb0, rb1);
    wmma_tiles(As[cur], Bs[cur], lane, mt, nh, acc);
    if (kb + 1 < NT) {
      *(bf16x8*)(&As[cur^1][ar][ach])  = ra;
      *(bf16x8*)(&Bs[cur^1][br0][bch]) = rb0;
      *(bf16x8*)(&Bs[cur^1][br1][bch]) = rb1;
    }
    __syncthreads();
  }
  // stash 64x128 gate pre-activations so threads can gather (i,f,g,o) quads
#pragma unroll
  for (int nt = 0; nt < 4; ++nt)
#pragma unroll
    for (int r = 0; r < 8; ++r)
      sg[mt*16 + ((lane >> 4) << 3) + r][nh*64 + nt*16 + (lane & 15)] = acc[nt][r];
  __syncthreads();

  const int ublk = nblk >> 2;   // 32 hidden units per block
  const size_t growbase = ((size_t)time * BB) * GDIM;
  for (int i = tid; i < 64*32; i += 256) {
    int m = i >> 5, u = i & 31;
    int unit = ublk + u;
    if (EDGE && unit >= HH) continue;
    float4 g4 = *(const float4*)(G + growbase + (size_t)m * GDIM + nblk + u*4);
    float4 s4 = *(const float4*)&sg[m][u*4];
    float pi = s4.x + g4.x;
    float pf = s4.y + g4.y;
    float pg = s4.z + g4.z;
    float po = s4.w + g4.w;
    float ig = 1.0f / (1.0f + __expf(-pi));
    float fg = 1.0f / (1.0f + __expf(-pf));
    float gg = tanhf(pg);
    float og = 1.0f / (1.0f + __expf(-po));
    float cn = fg * cst[m*HH + unit] + ig * gg;
    cst[m*HH + unit] = cn;
    float h = og * tanhf(cn);
    hnext[m*HH + unit] = (__bf16)h;
    hlayer[((size_t)time*BB + m) * (2*HH) + dir*HH + unit] = h;
  }
}

// ---------------------------------------------------------------------------
// Weight conversion f32 -> bf16. mode 0: copy [rows,cols]; mode 1: gate-permute
// rows (dst row j <- src row (j&3)*H + (j>>2)); mode 2: transpose (src [cols,rows]).
// ---------------------------------------------------------------------------
__global__ void convert_w_kernel(const float* __restrict__ src,
                                 __bf16* __restrict__ dst,
                                 int rows, int cols, int mode)
{
  int i = blockIdx.x * blockDim.x + threadIdx.x;
  if (i >= rows * cols) return;
  int j = i / cols, k = i - j * cols;
  float v;
  if (mode == 1) {
    int sr = (j & 3) * HH + (j >> 2);
    v = src[(size_t)sr * cols + k];
  } else if (mode == 2) {
    v = src[(size_t)k * rows + j];
  } else {
    v = src[(size_t)j * cols + k];
  }
  dst[(size_t)j * cols + k] = (__bf16)v;
}

__global__ void bias_perm_kernel(const float* __restrict__ s, float* __restrict__ d)
{
  int j = blockIdx.x * blockDim.x + threadIdx.x;
  if (j < GDIM) d[j] = s[(j & 3) * HH + (j >> 2)];
}

__global__ void embed_kernel(const int* __restrict__ us,
                             const float* __restrict__ emb,
                             __bf16* __restrict__ xb)
{
  int g = blockIdx.x * blockDim.x + threadIdx.x;
  if (g >= NTOK * EE) return;
  int n = g >> 8;          // token row (t*64+m)
  int e = g & 255;
  int tt = n >> 6, m = n & 63;
  int tok = us[m * TT + tt];
  xb[(size_t)n * EE + e] = (__bf16)emb[(size_t)tok * EE + e];
}

__global__ void cvt_f32_bf16_kernel(const float* __restrict__ s,
                                    __bf16* __restrict__ d, int n)
{
  int i = blockIdx.x * blockDim.x + threadIdx.x;
  if (i < n) d[i] = (__bf16)s[i];
}

__global__ void fill_f32_kernel(float* p, int n, float v)
{
  int i = blockIdx.x * blockDim.x + threadIdx.x;
  if (i < n) p[i] = v;
}

__global__ void fill_bf16_kernel(__bf16* p, int n)
{
  int i = blockIdx.x * blockDim.x + threadIdx.x;
  if (i < n) p[i] = (__bf16)0.0f;
}

// emissions[b,t,l] = bm2[l] + sum_k mid[n,k]*W2[k,l]   (L=4, K=600)
__global__ void mlp2_kernel(const float* __restrict__ mid,
                            const float* __restrict__ W2,
                            const float* __restrict__ bm2,
                            float* __restrict__ emis)
{
  int g = blockIdx.x * blockDim.x + threadIdx.x;
  if (g >= NTOK * 4) return;
  int n = g >> 2, l = g & 3;
  int t = n >> 6, m = n & 63;
  float s = bm2[l];
  const float* mr = mid + (size_t)n * 600;
  for (int k = 0; k < 600; ++k) s += mr[k] * W2[k*4 + l];
  emis[((size_t)m * TT + t) * 4 + l] = s;
}

// CRF log-likelihood (thread 0) + Viterbi decode (thread 1), one block per batch
__global__ void crf_kernel(const float* __restrict__ emis,
                           const int* __restrict__ ls,
                           const float* __restrict__ trans,
                           const float* __restrict__ startt,
                           const float* __restrict__ endt,
                           float* __restrict__ out)
{
  __shared__ short bp[TT][4];
  int m = blockIdx.x;
  const float* e = emis + (size_t)m * TT * 4;

  if (threadIdx.x == 0) {
    float alpha[4], na[4];
    for (int j = 0; j < 4; ++j) alpha[j] = startt[j] + e[j];
    for (int t = 1; t < TT; ++t) {
      for (int j = 0; j < 4; ++j) {
        float mx = -1e30f;
        for (int i = 0; i < 4; ++i) mx = fmaxf(mx, alpha[i] + trans[i*4 + j]);
        float sum = 0.0f;
        for (int i = 0; i < 4; ++i) sum += __expf(alpha[i] + trans[i*4 + j] - mx);
        na[j] = mx + __logf(sum) + e[t*4 + j];
      }
      for (int j = 0; j < 4; ++j) alpha[j] = na[j];
    }
    float mx = -1e30f;
    for (int j = 0; j < 4; ++j) mx = fmaxf(mx, alpha[j] + endt[j]);
    float sum = 0.0f;
    for (int j = 0; j < 4; ++j) sum += __expf(alpha[j] + endt[j] - mx);
    float den = mx + __logf(sum);

    const int* tg = ls + m * TT;
    float num = startt[tg[0]] + endt[tg[TT-1]];
    for (int t = 0; t < TT; ++t) num += e[t*4 + tg[t]];
    for (int t = 0; t + 1 < TT; ++t) num += trans[tg[t]*4 + tg[t+1]];
    atomicAdd(out, -(num - den) / (float)BB);
  } else if (threadIdx.x == 1) {
    float alpha[4], na[4];
    for (int j = 0; j < 4; ++j) alpha[j] = startt[j] + e[j];
    for (int t = 1; t < TT; ++t) {
      for (int j = 0; j < 4; ++j) {
        float best = -1e30f; int bi = 0;
        for (int i = 0; i < 4; ++i) {
          float s = alpha[i] + trans[i*4 + j];
          if (s > best) { best = s; bi = i; }
        }
        na[j] = best + e[t*4 + j];
        bp[t][j] = (short)bi;
      }
      for (int j = 0; j < 4; ++j) alpha[j] = na[j];
    }
    float best = -1e30f; int cur = 0;
    for (int j = 0; j < 4; ++j) {
      float s = alpha[j] + endt[j];
      if (s > best) { best = s; cur = j; }
    }
    for (int t = TT - 1; t >= 1; --t) {
      out[1 + m*TT + t] = (float)cur;
      cur = bp[t][cur];
    }
    out[1 + m*TT + 0] = (float)cur;
  }
}

// ---------------------------------------------------------------------------
extern "C" void kernel_launch(void* const* d_in, const int* in_sizes, int n_in,
                              void* d_out, int out_size, void* d_ws, size_t ws_size,
                              hipStream_t stream)
{
  (void)in_sizes; (void)n_in; (void)out_size; (void)ws_size;
  const int*   us    = (const int*)d_in[0];
  const int*   ls    = (const int*)d_in[1];
  const float* emb   = (const float*)d_in[2];
  const float* WihL[2][2] = {{(const float*)d_in[3],  (const float*)d_in[6]},
                             {(const float*)d_in[9],  (const float*)d_in[12]}};
  const float* WhhL[2][2] = {{(const float*)d_in[4],  (const float*)d_in[7]},
                             {(const float*)d_in[10], (const float*)d_in[13]}};
  const float* bL[2][2]   = {{(const float*)d_in[5],  (const float*)d_in[8]},
                             {(const float*)d_in[11], (const float*)d_in[14]}};
  const float* W1    = (const float*)d_in[15];
  const float* bm1   = (const float*)d_in[16];
  const float* W2    = (const float*)d_in[17];
  const float* bm2   = (const float*)d_in[18];
  const float* trans = (const float*)d_in[19];
  const float* st    = (const float*)d_in[20];
  const float* en    = (const float*)d_in[21];
  float* out = (float*)d_out;

  char* ws = (char*)d_ws;
  size_t off = 0;
  auto alloc = [&](size_t bytes) -> void* {
    void* p = ws + off;
    off = (off + bytes + 255) & ~(size_t)255;
    return p;
  };

  __bf16* xb   = (__bf16*)alloc((size_t)NTOK * EE * 2);
  float*  Gf   = (float*) alloc((size_t)NTOK * GDIM * 4);
  float*  Gb   = (float*) alloc((size_t)NTOK * GDIM * 4);
  float*  h0   = (float*) alloc((size_t)NTOK * 2*HH * 4);
  float*  h1   = (float*) alloc((size_t)NTOK * 2*HH * 4);
  __bf16* hb16 = (__bf16*)alloc((size_t)NTOK * 2*HH * 2);
  float*  mid  = (float*) alloc((size_t)NTOK * 600 * 4);
  float*  emis = (float*) alloc((size_t)BB * TT * 4 * 4);
  __bf16* wWih[2]  = {(__bf16*)alloc((size_t)GDIM*1200*2), (__bf16*)alloc((size_t)GDIM*1200*2)};
  __bf16* wWhh[2]  = {(__bf16*)alloc((size_t)GDIM*HH*2),   (__bf16*)alloc((size_t)GDIM*HH*2)};
  __bf16* wW1T     = (__bf16*)alloc((size_t)600*1200*2);
  float*  biasP[2] = {(float*)alloc(GDIM*4), (float*)alloc(GDIM*4)};
  __bf16* hp[2][2] = {{(__bf16*)alloc(BB*HH*2), (__bf16*)alloc(BB*HH*2)},
                      {(__bf16*)alloc(BB*HH*2), (__bf16*)alloc(BB*HH*2)}};
  float*  cws[2]   = {(float*)alloc(BB*HH*4), (float*)alloc(BB*HH*4)};

  // full blocks + one EDGE launch for the partial N tile
  auto launch_gemm = [&](const __bf16* A, const __bf16* Bw, const float* bias,
                         float* C, int K, int Ncols) {
    int nf = Ncols / 128, tail = Ncols % 128;
    if (K == EE) {
      if (nf)   gemm_bf16_t<EE, false><<<dim3(NTOK/64, nf), 256, 0, stream>>>(A, Bw, bias, C, Ncols, 0);
      if (tail) gemm_bf16_t<EE, true ><<<dim3(NTOK/64, 1 ), 256, 0, stream>>>(A, Bw, bias, C, Ncols, nf*128);
    } else {  // K == 1200
      if (nf)   gemm_bf16_t<1200, false><<<dim3(NTOK/64, nf), 256, 0, stream>>>(A, Bw, bias, C, Ncols, 0);
      if (tail) gemm_bf16_t<1200, true ><<<dim3(NTOK/64, 1 ), 256, 0, stream>>>(A, Bw, bias, C, Ncols, nf*128);
    }
  };

  // loss accumulator must start at 0 (harness poisons d_out)
  fill_f32_kernel<<<1, 32, 0, stream>>>(out, 1, 0.0f);

  // embedding lookup -> [T,B,E] bf16
  embed_kernel<<<(NTOK*EE + 255)/256, 256, 0, stream>>>(us, emb, xb);

  const int   KinL[2]   = {EE, 2*HH};
  const __bf16* AinL[2] = {xb, hb16};
  float* hlay[2] = {h0, h1};

  for (int l = 0; l < 2; ++l) {
    if (l == 1)  // layer-1 input = bf16(h0)
      cvt_f32_bf16_kernel<<<(NTOK*2*HH + 255)/256, 256, 0, stream>>>(h0, hb16, NTOK*2*HH);

    const int Kin = KinL[l];
    for (int d = 0; d < 2; ++d) {
      convert_w_kernel<<<(GDIM*Kin + 255)/256, 256, 0, stream>>>(WihL[l][d], wWih[d], GDIM, Kin, 1);
      convert_w_kernel<<<(GDIM*HH + 255)/256, 256, 0, stream>>>(WhhL[l][d], wWhh[d], GDIM, HH, 1);
      bias_perm_kernel<<<(GDIM + 255)/256, 256, 0, stream>>>(bL[l][d], biasP[d]);
    }
    // G = x @ Wih^T + b (gate-permuted), all 32768 tokens at once
    launch_gemm(AinL[l], wWih[0], biasP[0], Gf, Kin, GDIM);
    launch_gemm(AinL[l], wWih[1], biasP[1], Gb, Kin, GDIM);

    // zero recurrent state
    for (int d = 0; d < 2; ++d) {
      fill_f32_kernel<<<(BB*HH + 255)/256, 256, 0, stream>>>(cws[d], BB*HH, 0.0f);
      fill_bf16_kernel<<<(BB*HH + 255)/256, 256, 0, stream>>>(hp[d][0], BB*HH);
      fill_bf16_kernel<<<(BB*HH + 255)/256, 256, 0, stream>>>(hp[d][1], BB*HH);
    }
    // sequential recurrence: fwd runs t, bwd runs T-1-t, fused per launch
    const int nfull = GDIM / 128;            // 18 full blocks, 1 edge block
    for (int t = 0; t < TT; ++t) {
      int p = t & 1;
      lstm_step_t<false><<<dim3(nfull, 2), 256, 0, stream>>>(
          hp[0][p], hp[1][p], hp[0][p^1], hp[1][p^1],
          wWhh[0], wWhh[1], Gf, Gb, cws[0], cws[1], hlay[l], t, 0);
      lstm_step_t<true><<<dim3(1, 2), 256, 0, stream>>>(
          hp[0][p], hp[1][p], hp[0][p^1], hp[1][p^1],
          wWhh[0], wWhh[1], Gf, Gb, cws[0], cws[1], hlay[l], t, nfull*128);
    }
  }

  // MLP: mid = h1 @ W1 + bm1  (W1 [1200,600] -> transpose to [600,1200] bf16)
  cvt_f32_bf16_kernel<<<(NTOK*2*HH + 255)/256, 256, 0, stream>>>(h1, hb16, NTOK*2*HH);
  convert_w_kernel<<<(600*1200 + 255)/256, 256, 0, stream>>>(W1, wW1T, 600, 1200, 2);
  launch_gemm(hb16, wW1T, bm1, mid, 1200, 600);

  // emissions = mid @ W2 + bm2  -> [B,T,L]
  mlp2_kernel<<<(NTOK*4 + 255)/256, 256, 0, stream>>>(mid, W2, bm2, emis);

  // CRF loss + Viterbi tags
  crf_kernel<<<BB, 32, 0, stream>>>(emis, ls, trans, st, en, out);
}
